// SlotAttention_61856118997109
// MI455X (gfx1250) — matrix operand order repaired and checked by reading, not compiled, not run on hardware
//
#include <hip/hip_runtime.h>
#include <hip/hip_bf16.h>

// ---------------------------------------------------------------------------
// Slot attention on MI455X (gfx1250, wave32).
// Heavy GEMMs (LN@Wk/Wv projection, q@k^T logits, attn@v updates) run on
// v_wmma_f32_16x16x32_bf16; k/v are stored bf16 in workspace to halve HBM
// traffic (memory-bound problem: ~470MB @ 23.3TB/s ~ 20us; math ~36 GFLOP
// is negligible at WMMA rates). k+v bf16 = 128MB also fits the 192MB L2, so
// the attention sweeps after projection are mostly L2-resident.
// The v tile needed only in the updates phase is streamed into LDS by the
// Tensor Data Mover (tensor_load_to_lds, TENSORcnt) concurrently with the
// logits WMMA + softmax phase.
// Tiny slot-side ops (16x256) are VALU kernels.
// ---------------------------------------------------------------------------

typedef __attribute__((ext_vector_type(16))) __bf16 v16bf;
typedef __attribute__((ext_vector_type(8)))  float  v8f;
typedef __attribute__((ext_vector_type(4)))  unsigned int v4u;
typedef __attribute__((ext_vector_type(8)))  int v8i;
typedef __attribute__((ext_vector_type(4)))  int v4i;

#define WMMA_BF16(a, b, c) \
  __builtin_amdgcn_wmma_f32_16x16x32_bf16(false, (a), false, (b), (short)0, (c), false, false)

#define N_TOK 131072
#define F_DIM 256
#define D_DIM 256
#define K_SLOT 16
#define NB 128                 // tokens per workgroup in attention kernels
#define GB (N_TOK / NB)        // 1024 workgroups

union FragU { v16bf v; uint4 u[2]; };

// ---------------------------------------------------------------------------
// Kernel 0: pack Wk, Wv ([F,D], row = input dim) into transposed bf16
// WkT[n][d] so WMMA B-fragments are contiguous 32B runs.
// ---------------------------------------------------------------------------
__global__ __launch_bounds__(256) void k_pack_w(
    const float* __restrict__ Wk, const float* __restrict__ Wv,
    __bf16* __restrict__ WkT, __bf16* __restrict__ WvT) {
  for (int i = blockIdx.x * 256 + threadIdx.x; i < 2 * 65536; i += gridDim.x * 256) {
    int which = i >> 16, e = i & 65535;
    int n = e & 255, d = e >> 8;
    const float* src = which ? Wv : Wk;
    __bf16* dst = which ? WvT : WkT;
    dst[n * 256 + d] = (__bf16)src[d * 256 + n];  // read coalesced in n
  }
}

// ---------------------------------------------------------------------------
// Kernel 1: fused LayerNorm(features) + k/v projection via WMMA bf16.
// 128 rows per workgroup; normalized bf16 tile staged in LDS (stride 264
// elements -> 16B-aligned, bank-conflict-free A-fragment loads).
// A layout (16-bit 16x32, contiguous k): lane l (m = l&15, half = l>>4) holds one 16-elem
// run at K = kc*32 + 16*half (B fragments use the same k order).
// ---------------------------------------------------------------------------
__global__ __launch_bounds__(256) void k_ln_project(
    const float* __restrict__ feat, const float* __restrict__ g,
    const float* __restrict__ b, const __bf16* __restrict__ WkT,
    const __bf16* __restrict__ WvT, __bf16* __restrict__ kbf,
    __bf16* __restrict__ vbf) {
  __shared__ __align__(16) __bf16 fL[128 * 264];
  const int t = threadIdx.x, wave = t >> 5, lane = t & 31;
  const int hf = lane >> 4, l15 = lane & 15;
  const int rowBase = blockIdx.x * NB;

  // --- LayerNorm: each wave normalizes 16 rows (lane covers 8 columns) ---
  for (int i = 0; i < 16; ++i) {
    int lr = wave * 16 + i;
    const float* rp = feat + (size_t)(rowBase + lr) * F_DIM + lane * 8;
    float4 x0 = *(const float4*)rp;
    float4 x1 = *(const float4*)(rp + 4);
    float s = x0.x + x0.y + x0.z + x0.w + x1.x + x1.y + x1.z + x1.w;
    float q = x0.x * x0.x + x0.y * x0.y + x0.z * x0.z + x0.w * x0.w +
              x1.x * x1.x + x1.y * x1.y + x1.z * x1.z + x1.w * x1.w;
    for (int off = 1; off < 32; off <<= 1) {
      s += __shfl_xor(s, off);
      q += __shfl_xor(q, off);
    }
    float mean = s * (1.f / 256.f);
    float var = q * (1.f / 256.f) - mean * mean;
    float rstd = rsqrtf(var + 1e-5f);
    float xv[8] = {x0.x, x0.y, x0.z, x0.w, x1.x, x1.y, x1.z, x1.w};
    int c0 = lane * 8;
#pragma unroll
    for (int j = 0; j < 8; ++j) {
      float nv = (xv[j] - mean) * rstd * g[c0 + j] + b[c0 + j];
      fL[lr * 264 + c0 + j] = (__bf16)nv;
    }
  }
  __syncthreads();

  // --- WMMA: wave w owns row-tile w; preload 8 A-fragments, reuse for all
  //     32 output tiles (16 for k, 16 for v). ---
  v16bf A[8];
#pragma unroll
  for (int kc = 0; kc < 8; ++kc) {
    FragU f;
    int roff = (wave * 16 + l15) * 264;
    int base0 = kc * 32 + 16 * hf;
    f.u[0] = *(const uint4*)&fL[roff + base0];
    f.u[1] = *(const uint4*)&fL[roff + base0 + 8];
    A[kc] = f.v;
  }
  for (int w2 = 0; w2 < 2; ++w2) {
    const __bf16* WT = w2 ? WvT : WkT;
    __bf16* dst = w2 ? vbf : kbf;
    for (int nt = 0; nt < 16; ++nt) {
      v8f acc = {};
      const __bf16* wrow = WT + (size_t)(nt * 16 + l15) * F_DIM;
#pragma unroll
      for (int kc = 0; kc < 8; ++kc) {
        v16bf B = *(const v16bf*)(wrow + kc * 32 + 16 * hf);  // 32B contiguous
        acc = WMMA_BF16(A[kc], B, acc);
      }
      size_t col = nt * 16 + l15;
#pragma unroll
      for (int r = 0; r < 8; ++r) {
        int gm = rowBase + wave * 16 + r + 8 * hf;  // C layout: lane=col, vgpr=row
        dst[(size_t)gm * D_DIM + col] = (__bf16)acc[r];
      }
    }
  }
}

// ---------------------------------------------------------------------------
// Kernel 2: s = LN(slots); q = (s @ Wq) * D^-0.5  (scale folded into q).
// Single workgroup, tiny (1M MAC).
// ---------------------------------------------------------------------------
__global__ __launch_bounds__(256) void k_slot_q(
    const float* __restrict__ slots, const float* __restrict__ g,
    const float* __restrict__ b, const float* __restrict__ Wq,
    float* __restrict__ qout) {
  __shared__ float sL[16 * 256];
  __shared__ float mu[16], rs[16];
  int t = threadIdx.x;
  for (int i = 0; i < 16; ++i) sL[t + i * 256] = slots[t + i * 256];
  __syncthreads();
  if (t < 16) {
    float s = 0.f, q = 0.f;
    for (int d = 0; d < 256; ++d) {
      float x = sL[t * 256 + d];
      s += x;
      q += x * x;
    }
    float m = s * (1.f / 256.f);
    mu[t] = m;
    rs[t] = rsqrtf(q * (1.f / 256.f) - m * m + 1e-5f);
  }
  __syncthreads();
  float gc = g[t], bc = b[t];
#pragma unroll
  for (int m = 0; m < 16; ++m)
    sL[m * 256 + t] = (sL[m * 256 + t] - mu[m]) * rs[m] * gc + bc;
  __syncthreads();
  float acc[16];
#pragma unroll
  for (int m = 0; m < 16; ++m) acc[m] = 0.f;
  for (int d = 0; d < 256; ++d) {
    float w = Wq[d * 256 + t];  // coalesced
#pragma unroll
    for (int m = 0; m < 16; ++m) acc[m] += sL[m * 256 + d] * w;
  }
#pragma unroll
  for (int m = 0; m < 16; ++m) qout[m * 256 + t] = acc[m] * 0.0625f;  // 256^-0.5
}

// ---------------------------------------------------------------------------
// Fragment helper: q A-fragments from f32 q buffer (scaled).
// ---------------------------------------------------------------------------
__device__ __forceinline__ void load_q_frags(const float* __restrict__ qbuf,
                                             int l15, int hf, v16bf A[8]) {
#pragma unroll
  for (int kc = 0; kc < 8; ++kc) {
    v16bf a;
#pragma unroll
    for (int e = 0; e < 16; ++e) {
      int kk = kc * 32 + 16 * hf + e;
      a[e] = (__bf16)qbuf[l15 * 256 + kk];
    }
    A[kc] = a;
  }
}

// ---------------------------------------------------------------------------
// Kernel 3: one attention sweep with accumulation.
//  - wave 0 issues a TDM tensor_load_to_lds for the 128x256 bf16 v tile;
//    the DMA overlaps the logits WMMA + softmax phase (which needs only
//    q and k), and is fenced with s_wait_tensorcnt + barrier before the
//    attn@v updates WMMAs.
//  - logits = q@k^T (WMMA), softmax over 16 slots per column (lane<->lane^16
//    exchange matches C-layout row split), attn row-sum partials (LDS atomics
//    -> 16 global atomics/block), updates += attn@v (WMMA, attn restaged
//    through LDS to form A-fragments).
// ---------------------------------------------------------------------------
__global__ __launch_bounds__(256) void k_attn_acc(
    const float* __restrict__ qbuf, const __bf16* __restrict__ kbf,
    const __bf16* __restrict__ vbf, float* __restrict__ updPart,
    float* __restrict__ rowsumG) {
  __shared__ __align__(16) __bf16 vL[NB * D_DIM];  // 64 KB, TDM destination
  __shared__ __align__(16) __bf16 aL[16 * 144];    // attn tile, padded stride
  __shared__ float rsL[16];
  const int t = threadIdx.x, wave = t >> 5, lane = t & 31;
  const int hf = lane >> 4, l15 = lane & 15;
  const size_t blk = blockIdx.x;

  if (t < 16) rsL[t] = 0.f;

  // --- TDM: async-load the v tile into LDS (issued by wave 0 only; the
  //     Tensor DMA ignores EXEC, so every issuing wave would duplicate it).
  if (wave == 0) {
    unsigned long long ga =
        (unsigned long long)(uintptr_t)(vbf + blk * (size_t)NB * D_DIM);
    unsigned int la = (unsigned int)(uintptr_t)(&vL[0]);  // LDS offset in addr[31:0]
    v4u g0;
    g0[0] = 1u;                                   // count=1, user descriptor
    g0[1] = la;                                   // lds_addr (bytes)
    g0[2] = (unsigned int)ga;                     // global_addr[31:0]
    g0[3] = (unsigned int)((ga >> 32) & 0x1FFFFFFu) | (2u << 30);  // [56:32], type=2
    v8i g1;
    g1[0] = 1 << 16;                              // data_size=1 (2B); no mask/pad
    g1[1] = (int)(256u << 16);                    // tensor_dim0 = 256
    g1[2] = 0;                                    // dim0 hi / dim1 lo (131072&0xFFFF=0)
    g1[3] = (int)((131072u >> 16) | (256u << 16));// tensor_dim1 hi, tile_dim0=256
    g1[4] = 128;                                  // tile_dim1=128, tile_dim2=0
    g1[5] = 256;                                  // tensor_dim0_stride = 256
    g1[6] = 0;
    g1[7] = 0;
    v4i gz = {0, 0, 0, 0};
#if __clang_major__ >= 23
    v8i gz8 = {0, 0, 0, 0, 0, 0, 0, 0};
    __builtin_amdgcn_tensor_load_to_lds(g0, g1, gz, gz, gz8, 0);
#else
    __builtin_amdgcn_tensor_load_to_lds(g0, g1, gz, gz, 0);
#endif
  }

  v16bf A[8];
  load_q_frags(qbuf, l15, hf, A);

  // logits for this wave's 16 tokens (overlaps the TDM transfer)
  v8f acc = {};
  const __bf16* kr = kbf + (blk * NB + wave * 16 + l15) * (size_t)D_DIM;
#pragma unroll
  for (int kc = 0; kc < 8; ++kc) {
    v16bf B = *(const v16bf*)(kr + kc * 32 + 16 * hf);
    acc = WMMA_BF16(A[kc], B, acc);
  }

  // softmax over the 16 slots of each column (8 rows here + 8 in lane^16)
  float vmax = acc[0];
#pragma unroll
  for (int r = 1; r < 8; ++r) vmax = fmaxf(vmax, acc[r]);
  vmax = fmaxf(vmax, __shfl_xor(vmax, 16));
  float e8[8], s = 0.f;
#pragma unroll
  for (int r = 0; r < 8; ++r) {
    e8[r] = expf(acc[r] - vmax);
    s += e8[r];
  }
  s += __shfl_xor(s, 16);
  float inv = 1.f / s;
#pragma unroll
  for (int r = 0; r < 8; ++r) e8[r] *= inv;  // attn values

  // row-sum partials: reduce each row over the 16 columns of this wave
#pragma unroll
  for (int r = 0; r < 8; ++r) {
    float p = e8[r];
    for (int off = 1; off < 16; off <<= 1) p += __shfl_xor(p, off);
    if (l15 == 0) atomicAdd(&rsL[r + 8 * hf], p);
  }

  // stage attn tile (bf16) for A-fragment consumption
  int n0 = wave * 16;
#pragma unroll
  for (int r = 0; r < 8; ++r)
    aL[(r + 8 * hf) * 144 + n0 + l15] = (__bf16)e8[r];

  // Fence the TDM transfer (no-op for non-issuing waves) and publish LDS.
  __builtin_amdgcn_s_wait_tensorcnt(0);
  __syncthreads();

  if (t < 16) atomicAdd(&rowsumG[t], rsL[t]);

  // updates[16x256] partial = attn[16x128] @ v[128x256]; wave owns 2 d-tiles
  for (int ti = 0; ti < 2; ++ti) {
    int dt = wave * 2 + ti;
    v8f ua = {};
#pragma unroll
    for (int nc = 0; nc < 4; ++nc) {
      FragU fa;
      int base = l15 * 144 + nc * 32 + 8 * hf;
      fa.u[0] = *(const uint4*)&aL[base];
      fa.u[1] = *(const uint4*)&aL[base + 16];
      v16bf B;
#pragma unroll
      for (int e = 0; e < 16; ++e)
        B[e] = vL[(nc * 32 + 8 * hf + (e & 7) + 16 * (e >> 3)) * 256 + dt * 16 + l15];
      ua = WMMA_BF16(fa.v, B, ua);
    }
    float* up = updPart + blk * 4096;
#pragma unroll
    for (int r = 0; r < 8; ++r)
      up[(r + 8 * hf) * 256 + dt * 16 + l15] = ua[r];
  }
}

// ---------------------------------------------------------------------------
// Kernel 4: reduce 1024 per-block update partials -> updAcc[16][256].
// ---------------------------------------------------------------------------
__global__ __launch_bounds__(256) void k_reduce_upd(
    const float* __restrict__ updPart, float* __restrict__ updAcc) {
  int t = blockIdx.x * 256 + threadIdx.x;  // 0..4095
  float s = 0.f;
  for (int b = 0; b < GB; ++b) s += updPart[(size_t)b * 4096 + t];
  updAcc[t] = s;
}

// ---------------------------------------------------------------------------
// Kernel 5: GRU cell. updates = updAcc / (rowsum + 1.0).
// 768 threads: thread j computes gi[.][j], gh[.][j]; exchange via ws buffers.
// ---------------------------------------------------------------------------
__global__ __launch_bounds__(768) void k_gru(
    const float* __restrict__ updAcc, const float* __restrict__ rowsumG,
    const float* __restrict__ hprev, const float* __restrict__ wih,
    const float* __restrict__ whh, const float* __restrict__ bih,
    const float* __restrict__ bhh, float* __restrict__ gibuf,
    float* __restrict__ ghbuf, float* __restrict__ slots1) {
  __shared__ float uL[4096], hL[4096];
  int t = threadIdx.x;
  for (int idx = t; idx < 4096; idx += 768) {
    int m = idx >> 8;
    uL[idx] = updAcc[idx] / (rowsumG[m] + 1.0f);  // EPSILON = 1.0
    hL[idx] = hprev[idx];
  }
  __syncthreads();
  float gi[16], gh[16];
#pragma unroll
  for (int m = 0; m < 16; ++m) gi[m] = gh[m] = 0.f;
  const float* wi = wih + (size_t)t * 256;
  const float* wh = whh + (size_t)t * 256;
  for (int d = 0; d < 256; ++d) {
    float a = wi[d], c = wh[d];
#pragma unroll
    for (int m = 0; m < 16; ++m) {
      gi[m] += uL[m * 256 + d] * a;
      gh[m] += hL[m * 256 + d] * c;
    }
  }
  float bi = bih[t], bh = bhh[t];
#pragma unroll
  for (int m = 0; m < 16; ++m) {
    gibuf[m * 768 + t] = gi[m] + bi;
    ghbuf[m * 768 + t] = gh[m] + bh;
  }
  __syncthreads();
  if (t < 256) {
#pragma unroll
    for (int m = 0; m < 16; ++m) {
      float ir = gibuf[m * 768 + t];
      float iz = gibuf[m * 768 + 256 + t];
      float in = gibuf[m * 768 + 512 + t];
      float hr = ghbuf[m * 768 + t];
      float hz = ghbuf[m * 768 + 256 + t];
      float hn = ghbuf[m * 768 + 512 + t];
      float r = 1.f / (1.f + expf(-(ir + hr)));
      float z = 1.f / (1.f + expf(-(iz + hz)));
      float nn = tanhf(in + r * hn);
      slots1[m * 256 + t] = (1.f - z) * nn + z * hL[m * 256 + t];
    }
  }
}

// ---------------------------------------------------------------------------
// Kernel 6: slots += MLP(LN(slots)); straight-through re-attach, matching
// jnp evaluation order ((slots + slots_init) - slots_init).
// ---------------------------------------------------------------------------
__global__ __launch_bounds__(512) void k_mlp(
    const float* __restrict__ slots1, const float* __restrict__ g,
    const float* __restrict__ b, const float* __restrict__ w1,
    const float* __restrict__ b1, const float* __restrict__ w2,
    const float* __restrict__ b2, const float* __restrict__ slots_init,
    float* __restrict__ slots2, float* __restrict__ outSlots) {
  __shared__ float s1[4096], hln[4096], o1[16 * 512];
  __shared__ float mu[16], rs[16];
  int t = threadIdx.x;
  for (int idx = t; idx < 4096; idx += 512) s1[idx] = slots1[idx];
  __syncthreads();
  if (t < 16) {
    float s = 0.f, q = 0.f;
    for (int d = 0; d < 256; ++d) {
      float x = s1[t * 256 + d];
      s += x;
      q += x * x;
    }
    float m = s * (1.f / 256.f);
    mu[t] = m;
    rs[t] = rsqrtf(q * (1.f / 256.f) - m * m + 1e-5f);
  }
  __syncthreads();
  for (int idx = t; idx < 4096; idx += 512) {
    int m = idx >> 8, c = idx & 255;
    hln[idx] = (s1[idx] - mu[m]) * rs[m] * g[c] + b[c];
  }
  __syncthreads();
  {
    float acc[16];
#pragma unroll
    for (int m = 0; m < 16; ++m) acc[m] = 0.f;
    for (int d = 0; d < 256; ++d) {
      float w = w1[d * 512 + t];  // coalesced
#pragma unroll
      for (int m = 0; m < 16; ++m) acc[m] += hln[m * 256 + d] * w;
    }
    float bb = b1[t];
#pragma unroll
    for (int m = 0; m < 16; ++m) o1[m * 512 + t] = fmaxf(acc[m] + bb, 0.f);
  }
  __syncthreads();
  if (t < 256) {
    float acc[16];
#pragma unroll
    for (int m = 0; m < 16; ++m) acc[m] = 0.f;
    for (int r = 0; r < 512; ++r) {
      float w = w2[r * 256 + t];  // coalesced
#pragma unroll
      for (int m = 0; m < 16; ++m) acc[m] += o1[m * 512 + r] * w;
    }
    float bb = b2[t];
#pragma unroll
    for (int m = 0; m < 16; ++m) {
      float s2v = s1[m * 256 + t] + acc[m] + bb;
      float si = slots_init[m * 256 + t];
      float tmp = s2v + si;    // straight-through order:
      float res = tmp - si;    // (slots + slots_init) - slots_init
      slots2[m * 256 + t] = res;
      outSlots[m * 256 + t] = res;
    }
  }
}

// ---------------------------------------------------------------------------
// Kernel 7: final attention -> out[n][k] = softmax_k(q2 @ k_n). Transposed
// write through per-wave LDS tile for 64B-per-row coalesced stores.
// ---------------------------------------------------------------------------
__global__ __launch_bounds__(256) void k_final_attn(
    const float* __restrict__ qbuf, const __bf16* __restrict__ kbf,
    float* __restrict__ attnOut) {
  __shared__ float tL[8][16][17];
  const int t = threadIdx.x, wave = t >> 5, lane = t & 31;
  const int hf = lane >> 4, l15 = lane & 15;
  const size_t blk = blockIdx.x;

  v16bf A[8];
  load_q_frags(qbuf, l15, hf, A);

  v8f acc = {};
  const __bf16* kr = kbf + (blk * NB + wave * 16 + l15) * (size_t)D_DIM;
#pragma unroll
  for (int kc = 0; kc < 8; ++kc) {
    v16bf B = *(const v16bf*)(kr + kc * 32 + 16 * hf);
    acc = WMMA_BF16(A[kc], B, acc);
  }
  float vmax = acc[0];
#pragma unroll
  for (int r = 1; r < 8; ++r) vmax = fmaxf(vmax, acc[r]);
  vmax = fmaxf(vmax, __shfl_xor(vmax, 16));
  float e8[8], s = 0.f;
#pragma unroll
  for (int r = 0; r < 8; ++r) {
    e8[r] = expf(acc[r] - vmax);
    s += e8[r];
  }
  s += __shfl_xor(s, 16);
  float inv = 1.f / s;
#pragma unroll
  for (int r = 0; r < 8; ++r) tL[wave][l15][r + 8 * hf] = e8[r] * inv;
  __syncthreads();

  int row = lane >> 1, ch = (lane & 1) * 8;
  float4 f0 = {tL[wave][row][ch + 0], tL[wave][row][ch + 1],
               tL[wave][row][ch + 2], tL[wave][row][ch + 3]};
  float4 f1 = {tL[wave][row][ch + 4], tL[wave][row][ch + 5],
               tL[wave][row][ch + 6], tL[wave][row][ch + 7]};
  size_t gRow = blk * NB + wave * 16 + row;
  float* op = attnOut + gRow * 16 + ch;
  *(float4*)op = f0;
  *(float4*)(op + 4) = f1;
}

// ---------------------------------------------------------------------------
extern "C" void kernel_launch(void* const* d_in, const int* in_sizes, int n_in,
                              void* d_out, int out_size, void* d_ws,
                              size_t ws_size, hipStream_t stream) {
  const float* features = (const float*)d_in[0];
  const float* ln_f_g = (const float*)d_in[1];
  const float* ln_f_b = (const float*)d_in[2];
  const float* ln_s_g = (const float*)d_in[3];
  const float* ln_s_b = (const float*)d_in[4];
  const float* ln_m_g = (const float*)d_in[5];
  const float* ln_m_b = (const float*)d_in[6];
  const float* Wq = (const float*)d_in[7];
  const float* Wk = (const float*)d_in[8];
  const float* Wv = (const float*)d_in[9];
  const float* gru_wih = (const float*)d_in[10];
  const float* gru_whh = (const float*)d_in[11];
  const float* gru_bih = (const float*)d_in[12];
  const float* gru_bhh = (const float*)d_in[13];
  const float* mlp_w1 = (const float*)d_in[14];
  const float* mlp_b1 = (const float*)d_in[15];
  const float* mlp_w2 = (const float*)d_in[16];
  const float* mlp_b2 = (const float*)d_in[17];
  const float* slots_init = (const float*)d_in[18];
  // d_in[19] = num_iter (fixed at 2 in this problem: one update + final attn)

  char* ws = (char*)d_ws;
  size_t oK   = 0;                                  // k bf16  [N,D]
  size_t oV   = oK + (size_t)N_TOK * D_DIM * 2;     // v bf16  [N,D]
  size_t oWkT = oV + (size_t)N_TOK * D_DIM * 2;     // WkT bf16
  size_t oWvT = oWkT + 131072;
  size_t oQ   = oWvT + 131072;                      // q  f32 [16,256]
  size_t oQ2  = oQ + 16384;                         // q2 f32
  size_t oRS  = oQ2 + 16384;                        // rowsum f32[16]
  size_t oUA  = oRS + 256;                          // updAcc f32[4096]
  size_t oS1  = oUA + 16384;                        // slots after GRU
  size_t oS2  = oS1 + 16384;                        // slots after MLP
  size_t oGI  = oS2 + 16384;                        // gi f32[16,768]
  size_t oGH  = oGI + 49152;                        // gh f32[16,768]
  size_t oUP  = oGH + 49152;                        // updPart f32[1024,4096]

  __bf16* kbf = (__bf16*)(ws + oK);
  __bf16* vbf = (__bf16*)(ws + oV);
  __bf16* WkT = (__bf16*)(ws + oWkT);
  __bf16* WvT = (__bf16*)(ws + oWvT);
  float* qbuf = (float*)(ws + oQ);
  float* qbuf2 = (float*)(ws + oQ2);
  float* rowsum = (float*)(ws + oRS);
  float* updAcc = (float*)(ws + oUA);
  float* slots1 = (float*)(ws + oS1);
  float* slots2 = (float*)(ws + oS2);
  float* gibuf = (float*)(ws + oGI);
  float* ghbuf = (float*)(ws + oGH);
  float* updPart = (float*)(ws + oUP);

  float* out_slots = (float*)d_out;        // [16,256]
  float* out_attn = (float*)d_out + 4096;  // [N,16]

  hipMemsetAsync(rowsum, 0, 64, stream);
  k_pack_w<<<128, 256, 0, stream>>>(Wk, Wv, WkT, WvT);
  k_ln_project<<<GB, 256, 0, stream>>>(features, ln_f_g, ln_f_b, WkT, WvT,
                                       kbf, vbf);
  k_slot_q<<<1, 256, 0, stream>>>(slots_init, ln_s_g, ln_s_b, Wq, qbuf);
  k_attn_acc<<<GB, 256, 0, stream>>>(qbuf, kbf, vbf, updPart, rowsum);
  k_reduce_upd<<<16, 256, 0, stream>>>(updPart, updAcc);
  k_gru<<<1, 768, 0, stream>>>(updAcc, rowsum, slots_init, gru_wih, gru_whh,
                               gru_bih, gru_bhh, gibuf, ghbuf, slots1);
  k_mlp<<<1, 512, 0, stream>>>(slots1, ln_m_g, ln_m_b, mlp_w1, mlp_b1, mlp_w2,
                               mlp_b2, slots_init, slots2, out_slots);
  k_slot_q<<<1, 256, 0, stream>>>(slots2, ln_s_g, ln_s_b, Wq, qbuf2);
  k_final_attn<<<GB, 256, 0, stream>>>(qbuf2, kbf, out_attn);
}
